// SelfAttentionWithBias_60756607369319
// MI455X (gfx1250) — matrix operand order repaired and checked by reading, not compiled, not run on hardware
//
#include <hip/hip_runtime.h>
#include <hip/hip_bf16.h>
#include <math.h>

// Problem constants (reference: B=8, E=1024, D=256, H=8, HD=32, NB=6)
#define BB   8
#define EE   1024
#define DDIM 256
#define HH   8
#define HDIM 32
#define FFD  1024          // 4*D
#define BE   (BB * EE)     // 8192 rows

typedef __bf16 bf16;
typedef __attribute__((ext_vector_type(16))) __bf16 v16bf;
typedef __attribute__((ext_vector_type(8)))  __bf16 v8bf;
typedef __attribute__((ext_vector_type(8)))  float  v8f;

// ---------------------------------------------------------------------------
// WMMA fragment loaders (contiguous k = 16*half + s in both A and B, wave32).
// All loads are contiguous 16B/32B per lane -> global_load_b128 / ds_load_b128.
// ---------------------------------------------------------------------------

// A-matrix 16x32 bf16 from row-major source, leading dim `lda` (elements).
// lane&15 = row M; per lane: 16B at K=hf*16 and 16B at K=hf*16+8.
static __device__ __forceinline__ v16bf load_a_frag(const bf16* base, int lda, int lane) {
  const int m  = lane & 15;
  const int hf = (lane >> 4) & 1;
  const bf16* r = base + (size_t)m * lda + hf * 16;
  v8bf lo = *(const v8bf*)r;
  v8bf hi = *(const v8bf*)(r + 8);
  return __builtin_shufflevector(lo, hi, 0,1,2,3,4,5,6,7,8,9,10,11,12,13,14,15);
}

// B-matrix 32x16 from a TRANSPOSED source Bt[N][K] (row-major, ld = K):
// logical B[k][n] = Bt[n][k]; per lane a single contiguous 32B read.
static __device__ __forceinline__ v16bf load_bt_frag(const bf16* base, int ld, int lane) {
  const int n  = lane & 15;
  const int hf = (lane >> 4) & 1;
  const bf16* p = base + (size_t)n * ld + hf * 16;
  return *(const v16bf*)p;
}

// ---------------------------------------------------------------------------
// fp32 -> bf16 conversion (plain, and transposing for B operands)
// ---------------------------------------------------------------------------
__global__ void f2bf_kernel(const float* __restrict__ in, bf16* __restrict__ out, size_t n) {
  size_t i = (size_t)blockIdx.x * blockDim.x + threadIdx.x;
  if (i < n) out[i] = (bf16)in[i];
}

// in: row-major [K,N] fp32 -> out: row-major [N,K] bf16 (i.e., W^T)
__global__ void f2bf_t_kernel(const float* __restrict__ in, bf16* __restrict__ out, int K, int N) {
  int i = blockIdx.x * blockDim.x + threadIdx.x;
  if (i < K * N) {
    int k = i / N, n = i - k * N;
    out[(size_t)n * K + k] = (bf16)in[i];
  }
}

// ---------------------------------------------------------------------------
// Fused QKV projection: one A-fragment feeds three WMMA accumulators.
// Q,K out: [B,H,E,HD] bf16;  V out: TRANSPOSED [B,H,HD,E] bf16.
// ---------------------------------------------------------------------------
__global__ __launch_bounds__(256) void qkv_kernel(
    const bf16* __restrict__ xb,
    const bf16* __restrict__ wqt, const bf16* __restrict__ wkt, const bf16* __restrict__ wvt,
    const float* __restrict__ bq, const float* __restrict__ bk, const float* __restrict__ bv,
    bf16* __restrict__ qo, bf16* __restrict__ ko, bf16* __restrict__ vto) {
  const int lane = threadIdx.x & 31;
  const int wid  = (blockIdx.x * blockDim.x + threadIdx.x) >> 5;
  const int ntn  = DDIM >> 4;                 // 16 N-tiles
  const int mt   = wid / ntn, nt = wid - mt * ntn;
  if (mt * 16 >= BE) return;
  const bf16* At = xb + (size_t)mt * 16 * DDIM;
  const int n0 = nt * 16;
  v8f aq = {}, ak = {}, av = {};
  for (int k0 = 0; k0 < DDIM; k0 += 32) {
    __builtin_prefetch(At + k0 + 64, 0, 1);
    v16bf a  = load_a_frag(At + k0, DDIM, lane);
    v16bf b0 = load_bt_frag(wqt + (size_t)n0 * DDIM + k0, DDIM, lane);
    aq = __builtin_amdgcn_wmma_f32_16x16x32_bf16(false, a, false, b0, (short)0, aq, false, false);
    v16bf b1 = load_bt_frag(wkt + (size_t)n0 * DDIM + k0, DDIM, lane);
    ak = __builtin_amdgcn_wmma_f32_16x16x32_bf16(false, a, false, b1, (short)0, ak, false, false);
    v16bf b2 = load_bt_frag(wvt + (size_t)n0 * DDIM + k0, DDIM, lane);
    av = __builtin_amdgcn_wmma_f32_16x16x32_bf16(false, a, false, b2, (short)0, av, false, false);
  }
  const int n  = lane & 15;
  const int hf = (lane >> 4) & 1;
  const int col = n0 + n;
  const int h = col >> 5, hd = col & 31;      // HD = 32
  const float bqv = bq[col], bkv = bk[col], bvv = bv[col];
#pragma unroll
  for (int r = 0; r < 8; ++r) {
    const int g = mt * 16 + hf * 8 + r;       // row in [0, BE)
    const int b = g >> 10, e = g & (EE - 1);  // EE = 1024
    const size_t bh = (size_t)b * HH + h;
    const size_t o  = (bh * EE + e) * HDIM + hd;
    qo[o] = (bf16)(aq[r] + bqv);
    ko[o] = (bf16)(ak[r] + bkv);
    vto[(bh * HDIM + hd) * EE + e] = (bf16)(av[r] + bvv);   // V^T
  }
}

// ---------------------------------------------------------------------------
// Flash-style attention. grid = (B*H, E/64), block = 128 (4 waves),
// each wave owns a 16-query tile, streams 32 keys per iteration.
// K fragment: [E,HD] is per-lane contiguous for K^T. V fragment: from V^T.
// ---------------------------------------------------------------------------
__global__ __launch_bounds__(128) void attn_kernel(
    const bf16* __restrict__ qm, const bf16* __restrict__ km, const bf16* __restrict__ vtm,
    const int* __restrict__ rel, const unsigned char* __restrict__ kpm,
    const float* __restrict__ bias_emb, bf16* __restrict__ ctx) {
  __shared__ bf16 plds[4][16 * 32];           // P-tile transpose bounce, per wave
  const int lane = threadIdx.x & 31;
  const int w    = threadIdx.x >> 5;
  const int bh   = blockIdx.x;
  const int b = bh >> 3, h = bh & 7;          // H = 8
  const int q0 = (blockIdx.y * 4 + w) * 16;
  const bf16* qp  = qm  + (size_t)bh * EE * HDIM;
  const bf16* kp  = km  + (size_t)bh * EE * HDIM;
  const bf16* vtp = vtm + (size_t)bh * HDIM * EE;   // [HD, E]
  const int n  = lane & 15;
  const int hf = (lane >> 4) & 1;
  const float NEGINF = -__builtin_inff();

  // Q fragment: 16x32, K-dim == HD == 32 -> exactly one WMMA per score tile.
  v16bf aq = load_a_frag(qp + (size_t)q0 * HDIM, HDIM, lane);

  float mi[8], li[8];
  v8f acc0 = {}, acc1 = {};
#pragma unroll
  for (int r = 0; r < 8; ++r) { mi[r] = NEGINF; li[r] = 0.f; }
  const float scale = 0.17677669529663689f;   // HD^-0.5, HD=32
  bf16* pw = &plds[w][0];
  const v8f zero = {};

  for (int kb = 0; kb < EE; kb += 32) {
    // ---- scores: two 16x16 tiles (32 keys), one WMMA each ----
    v16bf bk0 = load_bt_frag(kp + (size_t)kb * HDIM, HDIM, lane);
    v16bf bk1 = load_bt_frag(kp + (size_t)(kb + 16) * HDIM, HDIM, lane);
    v8f s0v = __builtin_amdgcn_wmma_f32_16x16x32_bf16(false, aq, false, bk0, (short)0, zero, false, false);
    v8f s1v = __builtin_amdgcn_wmma_f32_16x16x32_bf16(false, aq, false, bk1, (short)0, zero, false, false);

    const int kp0 = kb + n, kp1 = kb + 16 + n;
    const bool msk0 = kpm[b * EE + kp0] != 0;
    const bool msk1 = kpm[b * EE + kp1] != 0;

    float s0[8], s1[8], tm[8];
#pragma unroll
    for (int r = 0; r < 8; ++r) {
      const int qi = q0 + hf * 8 + r;
      const size_t rbase = ((size_t)b * EE + qi) * EE;
      float a0 = s0v[r] * scale + bias_emb[rel[rbase + kp0] * HH + h];
      float a1 = s1v[r] * scale + bias_emb[rel[rbase + kp1] * HH + h];
      s0[r] = msk0 ? NEGINF : a0;
      s1[r] = msk1 ? NEGINF : a1;
      tm[r] = fmaxf(s0[r], s1[r]);
    }
    // row-max across the 16 lanes of this half (rows live per-half in C layout)
#pragma unroll
    for (int mm = 8; mm >= 1; mm >>= 1)
#pragma unroll
      for (int r = 0; r < 8; ++r) tm[r] = fmaxf(tm[r], __shfl_xor(tm[r], mm, 32));

    float rs[8], corr[8];
#pragma unroll
    for (int r = 0; r < 8; ++r) {
      float mn = fmaxf(mi[r], tm[r]);
      float ms = (mn == NEGINF) ? 0.f : mn;   // m_safe (fully-masked rows)
      float p0 = __expf(s0[r] - ms);          // exp(-inf) = 0 for masked keys
      float p1 = __expf(s1[r] - ms);
      corr[r]  = __expf(mi[r] - ms);
      mi[r] = mn;
      rs[r] = p0 + p1;
      pw[(hf * 8 + r) * 32 + n]      = (bf16)p0;
      pw[(hf * 8 + r) * 32 + 16 + n] = (bf16)p1;
    }
#pragma unroll
    for (int mm = 8; mm >= 1; mm >>= 1)
#pragma unroll
      for (int r = 0; r < 8; ++r) rs[r] += __shfl_xor(rs[r], mm, 32);
#pragma unroll
    for (int r = 0; r < 8; ++r) {
      li[r] = li[r] * corr[r] + rs[r];
      acc0[r] *= corr[r];
      acc1[r] *= corr[r];
    }

    // ---- P (C-layout) -> A-fragment via LDS; CDNA5 split DS counter ----
    asm volatile("s_wait_dscnt 0x0" ::: "memory");
    v16bf ap  = load_a_frag(pw, 32, lane);
    // V fragments from V^T: b[e] = V[k][d] = Vt[d][k]; contiguous 32B per lane.
    v16bf bv0 = load_bt_frag(vtp + kb,                     EE, lane);
    v16bf bv1 = load_bt_frag(vtp + (size_t)16 * EE + kb,   EE, lane);
    acc0 = __builtin_amdgcn_wmma_f32_16x16x32_bf16(false, ap, false, bv0, (short)0, acc0, false, false);
    acc1 = __builtin_amdgcn_wmma_f32_16x16x32_bf16(false, ap, false, bv1, (short)0, acc1, false, false);
  }

#pragma unroll
  for (int r = 0; r < 8; ++r) {
    float inv = li[r] > 0.f ? 1.0f / li[r] : 0.f;   // nan_to_num semantics
    const size_t row = (size_t)b * EE + q0 + hf * 8 + r;
    ctx[row * DDIM + h * HDIM + n]      = (bf16)(acc0[r] * inv);
    ctx[row * DDIM + h * HDIM + 16 + n] = (bf16)(acc1[r] * inv);
  }
}

// ---------------------------------------------------------------------------
// Generic bf16 WMMA GEMM, B pre-transposed (Bt[N][K]).
// Each wave computes a 16x64 tile: 4 accumulators share one A-fragment.
// ---------------------------------------------------------------------------
template <bool GELU, bool ADDRES, bool OUTBF>
__global__ __launch_bounds__(256) void gemm_bf16_kernel(
    const bf16* __restrict__ A, const bf16* __restrict__ Bt,
    const float* __restrict__ bias, const float* __restrict__ resid,
    void* __restrict__ out, int M, int N, int K) {
  const int lane = threadIdx.x & 31;
  const int wid  = (blockIdx.x * blockDim.x + threadIdx.x) >> 5;
  const int ng   = N >> 6;                       // 64-wide column groups
  const int mt   = wid / ng, gt = wid - mt * ng;
  if (mt * 16 >= M) return;
  const bf16* At = A + (size_t)mt * 16 * K;
  const int n0 = gt * 64;
  v8f acc[4] = {};
  for (int k0 = 0; k0 < K; k0 += 32) {
    __builtin_prefetch(At + k0 + 64, 0, 1);
    v16bf a = load_a_frag(At + k0, K, lane);
#pragma unroll
    for (int j = 0; j < 4; ++j) {
      v16bf b = load_bt_frag(Bt + (size_t)(n0 + 16 * j) * K + k0, K, lane);
      acc[j] = __builtin_amdgcn_wmma_f32_16x16x32_bf16(false, a, false, b, (short)0, acc[j], false, false);
    }
  }
  const int n  = lane & 15;
  const int hf = (lane >> 4) & 1;
#pragma unroll
  for (int j = 0; j < 4; ++j) {
    const int col = n0 + 16 * j + n;
    const float bv = bias[col];
#pragma unroll
    for (int r = 0; r < 8; ++r) {
      const int m = mt * 16 + hf * 8 + r;
      float v = acc[j][r] + bv;
      if (ADDRES) v += resid[(size_t)m * N + col];
      if (GELU)   v = 0.5f * v * (1.0f + erff(v * 0.70710678118654752f));  // exact GELU
      if (OUTBF) ((bf16*)out)[(size_t)m * N + col] = (bf16)v;
      else       ((float*)out)[(size_t)m * N + col] = v;
    }
  }
}

// ---------------------------------------------------------------------------
// Row LayerNorm over D=256 (block-per-row, 256 threads); optional bf16 copy.
// ---------------------------------------------------------------------------
template <bool EMITBF>
__global__ __launch_bounds__(256) void ln_kernel(
    const float* __restrict__ in, const float* __restrict__ gamma,
    const float* __restrict__ beta, float* __restrict__ outf, bf16* __restrict__ outb) {
  __shared__ float red[8];
  const int row = blockIdx.x;
  const int tid = threadIdx.x;
  const int lane = tid & 31, wid = tid >> 5;
  float v = in[(size_t)row * DDIM + tid];
  float s = v;
#pragma unroll
  for (int mm = 16; mm >= 1; mm >>= 1) s += __shfl_xor(s, mm, 32);
  if (lane == 0) red[wid] = s;
  __syncthreads();
  float mu = 0.f;
#pragma unroll
  for (int i = 0; i < 8; ++i) mu += red[i];
  mu *= (1.0f / DDIM);
  __syncthreads();
  float d = v - mu;
  s = d * d;
#pragma unroll
  for (int mm = 16; mm >= 1; mm >>= 1) s += __shfl_xor(s, mm, 32);
  if (lane == 0) red[wid] = s;
  __syncthreads();
  float var = 0.f;
#pragma unroll
  for (int i = 0; i < 8; ++i) var += red[i];
  var *= (1.0f / DDIM);
  float y = d * rsqrtf(var + 1e-5f) * gamma[tid] + beta[tid];
  outf[(size_t)row * DDIM + tid] = y;
  if (EMITBF) outb[(size_t)row * DDIM + tid] = (bf16)y;
}

// ---------------------------------------------------------------------------
// Host launcher
// ---------------------------------------------------------------------------
extern "C" void kernel_launch(void* const* d_in, const int* in_sizes, int n_in,
                              void* d_out, int out_size, void* d_ws, size_t ws_size,
                              hipStream_t stream) {
  (void)in_sizes; (void)n_in; (void)out_size; (void)ws_size;
  const float* x        = (const float*)d_in[0];
  const int*   rel      = (const int*)d_in[1];
  const unsigned char* kpm = (const unsigned char*)d_in[2];   // jax bool = 1 byte
  const float* wq  = (const float*)d_in[3];  const float* bq  = (const float*)d_in[4];
  const float* wk  = (const float*)d_in[5];  const float* bk  = (const float*)d_in[6];
  const float* wv  = (const float*)d_in[7];  const float* bv  = (const float*)d_in[8];
  const float* wo  = (const float*)d_in[9];  const float* bo  = (const float*)d_in[10];
  const float* bias_emb = (const float*)d_in[11];
  const float* g1  = (const float*)d_in[12]; const float* be1 = (const float*)d_in[13];
  const float* w1  = (const float*)d_in[14]; const float* b1f = (const float*)d_in[15];
  const float* w2  = (const float*)d_in[16]; const float* b2f = (const float*)d_in[17];
  const float* g2  = (const float*)d_in[18]; const float* be2 = (const float*)d_in[19];
  float* outp = (float*)d_out;

  // workspace carve-up (256B aligned)
  char* wsb = (char*)d_ws;
  size_t off = 0;
  auto take = [&](size_t bytes) -> char* {
    char* p = wsb + off; off += (bytes + 255) & ~(size_t)255; return p;
  };
  bf16* xb   = (bf16*)take((size_t)BE * DDIM * 2);
  bf16* wqt  = (bf16*)take((size_t)DDIM * DDIM * 2);   // transposed bf16 weights
  bf16* wkt  = (bf16*)take((size_t)DDIM * DDIM * 2);
  bf16* wvt  = (bf16*)take((size_t)DDIM * DDIM * 2);
  bf16* wot  = (bf16*)take((size_t)DDIM * DDIM * 2);
  bf16* w1t  = (bf16*)take((size_t)DDIM * FFD * 2);
  bf16* w2t  = (bf16*)take((size_t)FFD * DDIM * 2);
  bf16* qb   = (bf16*)take((size_t)BE * DDIM * 2);   // [B,H,E,HD]
  bf16* kb_  = (bf16*)take((size_t)BE * DDIM * 2);   // [B,H,E,HD]
  bf16* vtb  = (bf16*)take((size_t)BE * DDIM * 2);   // [B,H,HD,E]  (V^T)
  bf16* ctxb = (bf16*)take((size_t)BE * DDIM * 2);   // [BE, D]
  float* y1  = (float*)take((size_t)BE * DDIM * 4);  // pre-LN (reused for both)
  float* h1f = (float*)take((size_t)BE * DDIM * 4);
  bf16* h1b  = (bf16*)take((size_t)BE * DDIM * 2);
  bf16* ffb  = (bf16*)take((size_t)BE * FFD * 2);

  // conversions: x plain; all GEMM B-operands transposed
  f2bf_kernel<<<dim3((BE * DDIM + 255) / 256), dim3(256), 0, stream>>>(x, xb, (size_t)BE * DDIM);
  auto cvtT = [&](const float* src, bf16* dst, int K, int N) {
    f2bf_t_kernel<<<dim3((K * N + 255) / 256), dim3(256), 0, stream>>>(src, dst, K, N);
  };
  cvtT(wq, wqt, DDIM, DDIM);
  cvtT(wk, wkt, DDIM, DDIM);
  cvtT(wv, wvt, DDIM, DDIM);
  cvtT(wo, wot, DDIM, DDIM);
  cvtT(w1, w1t, DDIM, FFD);
  cvtT(w2, w2t, FFD, DDIM);

  // QKV: (BE/16)*(D/16) = 8192 wave-tiles, 8 waves/block -> 1024 blocks
  qkv_kernel<<<dim3(1024), dim3(256), 0, stream>>>(xb, wqt, wkt, wvt, bq, bk, bv, qb, kb_, vtb);

  // Attention: grid (B*H=64, E/64=16), 4 waves/block
  attn_kernel<<<dim3(BB * HH, EE / 64), dim3(128), 0, stream>>>(
      qb, kb_, vtb, rel, kpm, bias_emb, ctxb);

  // Out-proj + residual(x): M=8192 N=256 K=256 -> 2048 wave-tiles -> 256 blocks
  gemm_bf16_kernel<false, true, false><<<dim3(256), dim3(256), 0, stream>>>(
      ctxb, wot, bo, x, (void*)y1, BE, DDIM, DDIM);

  // LN1 -> h1 (fp32 + bf16)
  ln_kernel<true><<<dim3(BE), dim3(256), 0, stream>>>(y1, g1, be1, h1f, h1b);

  // FFN1 + GELU: M=8192 N=1024 K=256 -> 8192 wave-tiles -> 1024 blocks
  gemm_bf16_kernel<true, false, true><<<dim3(1024), dim3(256), 0, stream>>>(
      h1b, w1t, b1f, nullptr, (void*)ffb, BE, FFD, DDIM);

  // FFN2 + residual(h1): M=8192 N=256 K=1024 -> 2048 wave-tiles -> 256 blocks
  gemm_bf16_kernel<false, true, false><<<dim3(256), dim3(256), 0, stream>>>(
      ffb, w2t, b2f, h1f, (void*)y1, BE, DDIM, FFD);

  // LN2 -> d_out
  ln_kernel<false><<<dim3(BE), dim3(256), 0, stream>>>(y1, g2, be2, outp, nullptr);
}
